// RNN_ODE_Model_69698729279538
// MI455X (gfx1250) — matrix operand order repaired and checked by reading, not compiled, not run on hardware
//
#include <hip/hip_runtime.h>

typedef __attribute__((ext_vector_type(16))) _Float16 v16h;
typedef __attribute__((ext_vector_type(8)))  _Float16 v8h;
typedef __attribute__((ext_vector_type(8)))  float    v8f;
typedef _Float16 h16;

// ---- Tsit5 tableau ----
__device__ constexpr float TS_A[6][5] = {
  {0.f,0.f,0.f,0.f,0.f},
  {0.161f,0.f,0.f,0.f,0.f},
  {-0.008480655492356989f,0.335480655492357f,0.f,0.f,0.f},
  {2.8971530571054935f,-6.359448489975075f,4.3622954328695815f,0.f,0.f},
  {5.325864828439257f,-11.748883564062828f,7.4955393428898365f,-0.09249506636175525f,0.f},
  {5.86145544294642f,-12.92096931784711f,8.159367898576159f,-0.071584973281401f,-0.028269050394068383f}};
__device__ constexpr float TS_C[6] = {0.f,0.161f,0.327f,0.9f,0.9800255409045097f,1.0f};
__device__ constexpr float TS_B[6] = {0.09646076681806523f,0.01f,0.4798896504144996f,
                                      1.379008574103742f,-3.290069515436081f,2.324710524099774f};

// ---------- fragment loaders (two b128 loads each) ----------
__device__ inline v16h ld_fragA(const h16* p) {       // p = rowBase + k0 + kOffA
  v8h lo = *(const v8h*)p;
  v8h hi = *(const v8h*)(p + 16);
  return __builtin_shufflevector(lo, hi, 0,1,2,3,4,5,6,7,8,9,10,11,12,13,14,15);
}
__device__ inline v16h ld_fragB(const h16* p) {       // p = colBase + k0 + kOffB
  v8h lo = *(const v8h*)p;
  v8h hi = *(const v8h*)(p + 16);
  return __builtin_shufflevector(lo, hi, 0,1,2,3,4,5,6,7,8,9,10,11,12,13,14,15);
}

template<typename FA, typename FB>
__device__ inline v8f gemm_frag(int kSteps, FA fa, FB fb) {
  v8f acc = {0.f,0.f,0.f,0.f,0.f,0.f,0.f,0.f};
  for (int ks = 0; ks < kSteps; ++ks) {
    v16h a = fa(ks);
    v16h b = fb(ks);
    acc = __builtin_amdgcn_wmma_f32_16x16x32_f16(false, a, false, b, (short)0, acc, false, false);
  }
  return acc;
}

template<typename ST>
__device__ inline void store_tile(const v8f& acc, ST st) {
  int lane  = threadIdx.x & 31;
  int n     = lane & 15;
  int mBase = (lane < 16) ? 0 : 8;
#pragma unroll
  for (int r = 0; r < 8; ++r) st(mBase + r, n, acc[r]);
}

// ---------------- utility kernels ----------------
__global__ void k_cvt(const float* __restrict__ src, h16* __restrict__ dst, int n) {
  int i = blockIdx.x * 256 + threadIdx.x;
  if (i < n) dst[i] = (h16)src[i];
}

// pad rows from Kreal to Kpad with zeros
__global__ void k_cvt_pad(const float* __restrict__ src, h16* __restrict__ dst,
                          int rows, int Kreal, int Kpad) {
  int i = blockIdx.x * 256 + threadIdx.x;
  if (i >= rows * Kpad) return;
  int r = i / Kpad, k = i - r * Kpad;
  dst[i] = (k < Kreal) ? (h16)src[r * Kreal + k] : (h16)0.f;
}

// repack convT weights per parity class: wp[((pc*CoutP+oc)*Cin+ic)*4+t] = w[oc][ic][3-ky][3-kx]
__global__ void k_repackT(const float* __restrict__ w, h16* __restrict__ wp,
                          int Cout, int CoutP, int Cin) {
  int i = blockIdx.x * 256 + threadIdx.x;
  int total = 4 * CoutP * Cin * 4;
  if (i >= total) return;
  int t  = i & 3;
  int r  = i >> 2;
  int ic = r % Cin;  r /= Cin;
  int oc = r % CoutP;
  int pc = r / CoutP;
  int pY = pc >> 1, pX = pc & 1;
  int ky = pY + 2 * (t >> 1), kx = pX + 2 * (t & 1);
  float v = (oc < Cout) ? w[((oc * Cin + ic) * 4 + (3 - ky)) * 4 + (3 - kx)] : 0.f;
  wp[i] = (h16)v;
}

// video [B,10,64,64,3] f32 -> [320][3][64][64] f16
__global__ void k_video(const float* __restrict__ v, h16* __restrict__ dst, int n) {
  int i = blockIdx.x * 256 + threadIdx.x;
  if (i >= n) return;
  int w   = i & 63;
  int hh  = (i >> 6) & 63;
  int c   = (i >> 12) % 3;
  int img = (i >> 12) / 3;
  dst[i] = (h16)v[((img * 64 + hh) * 64 + w) * 3 + c];
}

// ---------------- encoder conv (implicit GEMM, k=4 s=2 p=1, relu) ----------------
template<int CIN, int HIN, int COUT, int HOUT, int KREAL, int KPAD>
__global__ void k_conv(const h16* __restrict__ x, const h16* __restrict__ w,
                       const float* __restrict__ bias, h16* __restrict__ y) {
  constexpr int HW = HOUT * HOUT;
  constexpr int NTN = COUT / 16;
  constexpr int NTM = (320 * HW) / 16;
  int lane = threadIdx.x & 31;
  int wave = (blockIdx.x * blockDim.x + threadIdx.x) >> 5;
  if (wave >= NTM * NTN) return;
  int tm = wave / NTN, tn = wave % NTN;
  int m0 = tm * 16, n0 = tn * 16;
  // per-lane A-row coordinates
  int mA   = m0 + (lane & 15);
  int imgA = mA / HW; int remA = mA - imgA * HW;
  int oyb  = (remA / HOUT) * 2 - 1, oxb = (remA % HOUT) * 2 - 1;
  const h16* xb = x + (size_t)imgA * CIN * HIN * HIN;
  int kOffA = (lane < 16) ? 0 : 8;
  const h16* pB = w + (size_t)(n0 + (lane & 15)) * KPAD + ((lane < 16) ? 0 : 8);
  auto fa = [&](int ks) -> v16h {
    v16h a;
#pragma unroll
    for (int j = 0; j < 16; ++j) {
      int k  = ks * 32 + kOffA + (j < 8 ? j : j + 8);
      int ic = k >> 4; int kk = k & 15;
      int iy = oyb + (kk >> 2), ix = oxb + (kk & 3);
      bool ok = (k < KREAL) & ((unsigned)iy < (unsigned)HIN) & ((unsigned)ix < (unsigned)HIN);
      int idx = (ic * HIN + iy) * HIN + ix;
      h16 v = xb[ok ? idx : 0];
      a[j] = ok ? v : (h16)0.f;
    }
    return a;
  };
  auto fb = [&](int ks) { return ld_fragB(pB + ks * 32); };
  v8f acc = gemm_frag(KPAD / 32, fa, fb);
  // lane writes 8 consecutive spatial positions of one channel -> b128 store
  int oc   = n0 + (lane & 15);
  int mS   = m0 + ((lane < 16) ? 0 : 8);
  int imgS = mS / HW; int remS = mS - imgS * HW;
  float bb = bias[oc];
  v8h ov;
#pragma unroll
  for (int r = 0; r < 8; ++r) {
    float o = acc[r] + bb;
    ov[r] = (h16)(o > 0.f ? o : 0.f);
  }
  *(v8h*)(y + ((size_t)(imgS * COUT + oc) * HW + remS)) = ov;
}

// ---------------- generic linear: out = A @ W^T + bias ----------------
__global__ void k_linear(const h16* __restrict__ A, const h16* __restrict__ W,
                         const float* __restrict__ bias, void* __restrict__ out,
                         int M, int K, int N, int outF16) {
  int lane = threadIdx.x & 31;
  int wave = (blockIdx.x * blockDim.x + threadIdx.x) >> 5;
  int nTilesN = N >> 4;
  if (wave >= (M >> 4) * nTilesN) return;
  int tm = wave / nTilesN, tn = wave % nTilesN;
  const h16* pA = A + (size_t)(tm * 16 + (lane & 15)) * K + ((lane < 16) ? 0 : 8);
  const h16* pB = W + (size_t)(tn * 16 + (lane & 15)) * K + ((lane < 16) ? 0 : 8);
  v8f acc = gemm_frag(K >> 5,
                      [&](int ks) { return ld_fragA(pA + ks * 32); },
                      [&](int ks) { return ld_fragB(pB + ks * 32); });
  store_tile(acc, [&](int r, int c, float v) {
    int m = tm * 16 + r, n = tn * 16 + c;
    float o = v + bias[n];
    if (outF16) ((h16*)out)[(size_t)m * N + n] = (h16)o;
    else        ((float*)out)[(size_t)m * N + n] = o;
  });
}

// ---------------- GRU recurrence + mu/lv/z0 (single workgroup, 8 waves) ----------------
__global__ void __launch_bounds__(256) k_gru(
    const float* __restrict__ ig, const h16* __restrict__ whh,
    const float* __restrict__ gbn,
    const h16* __restrict__ muw, const float* __restrict__ mub,
    const h16* __restrict__ lvw, const float* __restrict__ lvb,
    const float* __restrict__ eps,
    float* __restrict__ hg, float* __restrict__ muo, float* __restrict__ lvo,
    float* __restrict__ z0, h16* __restrict__ ztraj) {
  __shared__ float hS[32 * 256];
  __shared__ alignas(16) h16 hF[32 * 256];
  int tid = threadIdx.x, wave = tid >> 5, lane = tid & 31;
  int kOffA = (lane < 16) ? 0 : 8;
  int kOffB = (lane < 16) ? 0 : 8;
  for (int i = tid; i < 32 * 256; i += 256) { hS[i] = 0.f; hF[i] = (h16)0.f; }
  __syncthreads();
  for (int t = 0; t < 10; ++t) {
    for (int tile = wave; tile < 96; tile += 8) {   // hg = h @ whh^T  (32x256x768)
      int tm = tile / 48, tn = tile % 48;
      const h16* pA = hF + (tm * 16 + (lane & 15)) * 256 + kOffA;
      const h16* pB = whh + (size_t)(tn * 16 + (lane & 15)) * 256 + kOffB;
      v8f acc = gemm_frag(8, [&](int ks) { return ld_fragA(pA + ks * 32); },
                             [&](int ks) { return ld_fragB(pB + ks * 32); });
      store_tile(acc, [&](int r, int c, float v) { hg[(tm * 16 + r) * 768 + tn * 16 + c] = v; });
    }
    __threadfence(); __syncthreads();
    for (int i = tid; i < 32 * 256; i += 256) {
      int b = i >> 8, c = i & 255;
      const float* igr = ig + (b * 10 + t) * 768;
      float i0 = igr[c], i1 = igr[256 + c], i2 = igr[512 + c];
      float h0 = hg[b * 768 + c], h1 = hg[b * 768 + 256 + c], h2 = hg[b * 768 + 512 + c];
      float reset = 1.f / (1.f + __expf(-(i0 + h0)));
      float inp   = 1.f / (1.f + __expf(-(i1 + h1)));
      float nv    = tanhf(i2 + reset * (h2 + gbn[c]));
      float hn    = nv + inp * (hS[i] - nv);
      hS[i] = hn; hF[i] = (h16)hn;
    }
    __syncthreads();
  }
  for (int tile = wave; tile < 16; tile += 8) {     // mu & logvar (32x256x64 each)
    int mat = tile >> 3; int tt = tile & 7;
    int tm = tt >> 2, tn = tt & 3;
    const h16*   W = mat ? lvw : muw;
    const float* B = mat ? lvb : mub;
    float*       O = mat ? lvo : muo;
    const h16* pA = hF + (tm * 16 + (lane & 15)) * 256 + kOffA;
    const h16* pB = W + (size_t)(tn * 16 + (lane & 15)) * 256 + kOffB;
    v8f acc = gemm_frag(8, [&](int ks) { return ld_fragA(pA + ks * 32); },
                           [&](int ks) { return ld_fragB(pB + ks * 32); });
    store_tile(acc, [&](int r, int c, float v) { O[(tm * 16 + r) * 64 + tn * 16 + c] = v + B[tn * 16 + c]; });
  }
  __threadfence(); __syncthreads();
  for (int i = tid; i < 2048; i += 256) {
    float z = muo[i] + eps[i] * __expf(0.5f * lvo[i]);
    z0[i] = z;
    ztraj[i] = (h16)z;                              // z_traj[0]
  }
}

// ---------------- persistent Tsit5 neural-ODE (single workgroup, 8 waves) ----------------
__global__ void __launch_bounds__(256) k_ode(
    const float* __restrict__ z0,
    const h16* __restrict__ w1p, const float* __restrict__ b1,   // w1p: [256][96] padded
    const h16* __restrict__ w2, const float* __restrict__ b2,
    const h16* __restrict__ w3, const float* __restrict__ b3,
    const h16* __restrict__ w4, const float* __restrict__ b4,
    h16* __restrict__ ztraj) {
  __shared__ alignas(16) h16   yt[32 * 96];     // [z | t | pad] f16
  __shared__ alignas(16) h16   actA[32 * 256];
  __shared__ alignas(16) h16   actB[32 * 256];
  __shared__ float kout[32 * 64];
  int tid = threadIdx.x, wave = tid >> 5, lane = tid & 31;
  int kOffA = (lane < 16) ? 0 : 8;
  int kOffB = (lane < 16) ? 0 : 8;
  int base = tid * 8;
  float yreg[8], kreg[6][8];
#pragma unroll
  for (int j = 0; j < 8; ++j) yreg[j] = z0[base + j];

  for (int save = 0; save < 19; ++save) {
    for (int st = 0; st < 10; ++st) {
      float t0 = (float)(save * 10 + st) * 0.1f;
#pragma unroll
      for (int s = 0; s < 6; ++s) {
#pragma unroll
        for (int j = 0; j < 8; ++j) {
          float v = yreg[j];
#pragma unroll
          for (int jj = 0; jj < 5; ++jj)
            if (jj < s) v += 0.1f * TS_A[s][jj] * kreg[jj][j];
          int idx = base + j; int m = idx >> 6, c = idx & 63;
          yt[m * 96 + c] = (h16)v;
        }
        float ts = t0 + TS_C[s] * 0.1f;
#pragma unroll
        for (int p = 0; p < 4; ++p) {
          int idx = tid * 4 + p;                  // 1024 pad slots
          int m = idx >> 5; int cc = 64 + (idx & 31);
          yt[m * 96 + cc] = (cc == 64) ? (h16)ts : (h16)0.f;
        }
        __syncthreads();
        // L1: 32x96x256 (Kpad=96), softplus
        for (int tile = wave; tile < 32; tile += 8) {
          int tm = tile >> 4, tn = tile & 15;
          const h16* pA = yt + (tm * 16 + (lane & 15)) * 96 + kOffA;
          const h16* pB = w1p + (size_t)(tn * 16 + (lane & 15)) * 96 + kOffB;
          v8f acc = gemm_frag(3, [&](int ks) { return ld_fragA(pA + ks * 32); },
                                 [&](int ks) { return ld_fragB(pB + ks * 32); });
          store_tile(acc, [&](int r, int c, float v) {
            float x = v + b1[tn * 16 + c];
            float sp = (x > 0.f ? x : 0.f) + log1pf(__expf(-fabsf(x)));
            actA[(tm * 16 + r) * 256 + tn * 16 + c] = (h16)sp; });
        }
        __syncthreads();
        // L2
        for (int tile = wave; tile < 32; tile += 8) {
          int tm = tile >> 4, tn = tile & 15;
          const h16* pA = actA + (tm * 16 + (lane & 15)) * 256 + kOffA;
          const h16* pB = w2 + (size_t)(tn * 16 + (lane & 15)) * 256 + kOffB;
          v8f acc = gemm_frag(8, [&](int ks) { return ld_fragA(pA + ks * 32); },
                                 [&](int ks) { return ld_fragB(pB + ks * 32); });
          store_tile(acc, [&](int r, int c, float v) {
            float x = v + b2[tn * 16 + c];
            float sp = (x > 0.f ? x : 0.f) + log1pf(__expf(-fabsf(x)));
            actB[(tm * 16 + r) * 256 + tn * 16 + c] = (h16)sp; });
        }
        __syncthreads();
        // L3
        for (int tile = wave; tile < 32; tile += 8) {
          int tm = tile >> 4, tn = tile & 15;
          const h16* pA = actB + (tm * 16 + (lane & 15)) * 256 + kOffA;
          const h16* pB = w3 + (size_t)(tn * 16 + (lane & 15)) * 256 + kOffB;
          v8f acc = gemm_frag(8, [&](int ks) { return ld_fragA(pA + ks * 32); },
                                 [&](int ks) { return ld_fragB(pB + ks * 32); });
          store_tile(acc, [&](int r, int c, float v) {
            float x = v + b3[tn * 16 + c];
            float sp = (x > 0.f ? x : 0.f) + log1pf(__expf(-fabsf(x)));
            actA[(tm * 16 + r) * 256 + tn * 16 + c] = (h16)sp; });
        }
        __syncthreads();
        // L4: 32x256x64 -> kout
        for (int tile = wave; tile < 8; tile += 8) {
          int tm = tile >> 2, tn = tile & 3;
          const h16* pA = actA + (tm * 16 + (lane & 15)) * 256 + kOffA;
          const h16* pB = w4 + (size_t)(tn * 16 + (lane & 15)) * 256 + kOffB;
          v8f acc = gemm_frag(8, [&](int ks) { return ld_fragA(pA + ks * 32); },
                                 [&](int ks) { return ld_fragB(pB + ks * 32); });
          store_tile(acc, [&](int r, int c, float v) { kout[(tm * 16 + r) * 64 + tn * 16 + c] = v + b4[tn * 16 + c]; });
        }
        __syncthreads();
#pragma unroll
        for (int j = 0; j < 8; ++j) kreg[s][j] = kout[base + j];
      }
#pragma unroll
      for (int j = 0; j < 8; ++j) {
        float d = 0.f;
#pragma unroll
        for (int s = 0; s < 6; ++s) d += TS_B[s] * kreg[s][j];
        yreg[j] += 0.1f * d;
      }
    }
#pragma unroll
    for (int j = 0; j < 8; ++j) ztraj[(save + 1) * 2048 + base + j] = (h16)yreg[j];
  }
}

// ---------------- transposed conv (implicit GEMM per output-parity class) ----------------
template<int CIN, int HIN, int COUT, int COUTP, int HOUT, int LAST>
__global__ void k_convT(const h16* __restrict__ x, const h16* __restrict__ wp,
                        const float* __restrict__ bias, void* __restrict__ out) {
  constexpr int HO2 = HOUT / 2;
  constexpr int HW2 = HO2 * HO2;
  constexpr int K   = CIN * 4;
  constexpr int NTN = COUTP / 16;
  constexpr int NTM = (640 * HW2) / 16;
  int pclass = blockIdx.y;              // 0..3
  int pY = pclass >> 1, pX = pclass & 1;
  int lane = threadIdx.x & 31;
  int wave = (blockIdx.x * blockDim.x + threadIdx.x) >> 5;
  if (wave >= NTM * NTN) return;
  int tm = wave / NTN, tn = wave % NTN;
  // per-lane A-row coords
  int mA   = tm * 16 + (lane & 15);
  int imgA = mA / HW2; int remA = mA - imgA * HW2;
  int iyb  = remA / HO2 + pY - 1, ixb = remA % HO2 + pX - 1;
  const h16* xb = x + (size_t)imgA * CIN * HIN * HIN;
  int kOffA = (lane < 16) ? 0 : 8;
  const h16* pB = wp + (size_t)(pclass * COUTP + tn * 16 + (lane & 15)) * K + ((lane < 16) ? 0 : 8);
  auto fa = [&](int ks) -> v16h {
    v16h a;
#pragma unroll
    for (int j = 0; j < 16; ++j) {
      int k  = ks * 32 + kOffA + (j < 8 ? j : j + 8);
      int ic = k >> 2; int tky = (k >> 1) & 1, tkx = k & 1;
      int iy = iyb + tky, ix = ixb + tkx;
      bool ok = ((unsigned)iy < (unsigned)HIN) & ((unsigned)ix < (unsigned)HIN);
      int idx = (ic * HIN + iy) * HIN + ix;
      h16 v = xb[ok ? idx : 0];
      a[j] = ok ? v : (h16)0.f;
    }
    return a;
  };
  v8f acc = gemm_frag(K / 32, fa, [&](int ks) { return ld_fragB(pB + ks * 32); });
  store_tile(acc, [&](int r, int c, float v) {
    int oc = tn * 16 + c;
    if (oc >= COUT) return;
    int m = tm * 16 + r;
    int img = m / HW2; int rem = m - img * HW2;
    int oy = (rem / HO2) * 2 + pY, ox = (rem % HO2) * 2 + pX;
    float o = v + bias[oc];
    if (!LAST) {
      ((h16*)out)[((size_t)(img * COUT + oc) * HOUT + oy) * HOUT + ox] = (h16)(o > 0.f ? o : 0.f);
    } else {
      float sgm = 1.f / (1.f + __expf(-o));
      int b = img & 31, tt = img >> 5;   // img = t*32 + b
      ((float*)out)[(((size_t)(b * 20 + tt) * 64 + oy) * 64 + ox) * 3 + oc] = sgm;
    }
  });
}

__global__ void k_kl(const float* __restrict__ mu, const float* __restrict__ lv,
                     float* __restrict__ out) {
  int b = threadIdx.x;
  if (b >= 32) return;
  float s = 0.f;
  for (int i = 0; i < 64; ++i) {
    float m = mu[b * 64 + i], l = lv[b * 64 + i];
    s += 1.f + l - m * m - __expf(l);
  }
  out[b] = -0.5f * s;
}

// ---------------- host launcher ----------------
extern "C" void kernel_launch(void* const* d_in, const int* in_sizes, int n_in,
                              void* d_out, int out_size, void* d_ws, size_t ws_size,
                              hipStream_t stream) {
  const float* ref_videos = (const float*)d_in[0];
  const float* eps        = (const float*)d_in[1];
  const float* enc_w[4]   = {(const float*)d_in[2], (const float*)d_in[3], (const float*)d_in[4], (const float*)d_in[5]};
  const float* enc_b[4]   = {(const float*)d_in[6], (const float*)d_in[7], (const float*)d_in[8], (const float*)d_in[9]};
  const float* gru_wih = (const float*)d_in[10];
  const float* gru_whh = (const float*)d_in[11];
  const float* gru_b   = (const float*)d_in[12];
  const float* gru_bn  = (const float*)d_in[13];
  const float* mu_w = (const float*)d_in[14]; const float* mu_b = (const float*)d_in[15];
  const float* lv_w = (const float*)d_in[16]; const float* lv_b = (const float*)d_in[17];
  const float* vf_w[4] = {(const float*)d_in[18], (const float*)d_in[19], (const float*)d_in[20], (const float*)d_in[21]};
  const float* vf_b[4] = {(const float*)d_in[22], (const float*)d_in[23], (const float*)d_in[24], (const float*)d_in[25]};
  const float* decl_w = (const float*)d_in[26];
  const float* decl_b = (const float*)d_in[27];
  const float* dec_w[4] = {(const float*)d_in[28], (const float*)d_in[29], (const float*)d_in[30], (const float*)d_in[31]};
  const float* dec_b[4] = {(const float*)d_in[32], (const float*)d_in[33], (const float*)d_in[34], (const float*)d_in[35]};

  char* p = (char*)d_ws;
  auto alloc = [&](size_t bytes) -> void* {
    void* r = (void*)p; p += (bytes + 255) & ~(size_t)255; return r;
  };
  h16* x016   = (h16*)alloc((size_t)320*3*64*64*2);
  h16* a1     = (h16*)alloc((size_t)320*32*32*32*2);
  h16* a2     = (h16*)alloc((size_t)320*64*16*16*2);
  h16* a3     = (h16*)alloc((size_t)320*128*8*8*2);
  h16* feats  = (h16*)alloc((size_t)320*4096*2);
  // padded/repacked f16 weights
  h16* encw16[4]; int encwN[4] = {32*64 /*padded 48->64*/, 64*512, 128*1024, 256*2048};
  for (int i = 0; i < 4; ++i) encw16[i] = (h16*)alloc((size_t)encwN[i]*2);
  h16* wih16  = (h16*)alloc((size_t)768*4096*2);
  h16* whh16  = (h16*)alloc((size_t)768*256*2);
  h16* muw16  = (h16*)alloc((size_t)64*256*2);
  h16* lvw16  = (h16*)alloc((size_t)64*256*2);
  h16* vfw16[4]; int vfwN[4] = {256*96 /*padded 65->96*/, 256*256, 256*256, 64*256};
  for (int i = 0; i < 4; ++i) vfw16[i] = (h16*)alloc((size_t)vfwN[i]*2);
  h16* declw16 = (h16*)alloc((size_t)2048*64*2);
  h16* decw16[4]; int decwN[4] = {4*64*128*4, 4*32*64*4, 4*16*32*4, 4*16*16*4}; // repacked per class
  for (int i = 0; i < 4; ++i) decw16[i] = (h16*)alloc((size_t)decwN[i]*2);
  float* ig   = (float*)alloc((size_t)320*768*4);
  float* hg   = (float*)alloc((size_t)32*768*4);
  float* muo  = (float*)alloc((size_t)2048*4);
  float* lvo  = (float*)alloc((size_t)2048*4);
  float* z0   = (float*)alloc((size_t)2048*4);
  h16* ztraj  = (h16*)alloc((size_t)640*64*2);
  h16* d0     = (h16*)alloc((size_t)640*2048*2);
  h16* d1     = (h16*)alloc((size_t)640*64*64*2);
  h16* d2     = (h16*)alloc((size_t)640*32*256*2);
  h16* d3     = (h16*)alloc((size_t)640*16*1024*2);

  auto cvt = [&](const float* s, h16* d, int n) {
    k_cvt<<<(n + 255) / 256, 256, 0, stream>>>(s, d, n);
  };
  // enc layer1 padded 48->64, others direct
  k_cvt_pad<<<(32*64 + 255) / 256, 256, 0, stream>>>(enc_w[0], encw16[0], 32, 48, 64);
  cvt(enc_w[1], encw16[1], encwN[1]);
  cvt(enc_w[2], encw16[2], encwN[2]);
  cvt(enc_w[3], encw16[3], encwN[3]);
  cvt(gru_wih, wih16, 768*4096);
  cvt(gru_whh, whh16, 768*256);
  cvt(mu_w, muw16, 64*256);
  cvt(lv_w, lvw16, 64*256);
  k_cvt_pad<<<(256*96 + 255) / 256, 256, 0, stream>>>(vf_w[0], vfw16[0], 256, 65, 96);
  cvt(vf_w[1], vfw16[1], vfwN[1]);
  cvt(vf_w[2], vfw16[2], vfwN[2]);
  cvt(vf_w[3], vfw16[3], vfwN[3]);
  cvt(decl_w, declw16, 2048*64);
  k_repackT<<<(decwN[0] + 255) / 256, 256, 0, stream>>>(dec_w[0], decw16[0], 64, 64, 128);
  k_repackT<<<(decwN[1] + 255) / 256, 256, 0, stream>>>(dec_w[1], decw16[1], 32, 32, 64);
  k_repackT<<<(decwN[2] + 255) / 256, 256, 0, stream>>>(dec_w[2], decw16[2], 16, 16, 32);
  k_repackT<<<(decwN[3] + 255) / 256, 256, 0, stream>>>(dec_w[3], decw16[3],  3, 16, 16);

  { int n = 320*3*64*64; k_video<<<(n + 255) / 256, 256, 0, stream>>>(ref_videos, x016, n); }

  // encoder convs (implicit GEMM)
  { int tiles = (320*32*32/16)*(32/16);
    k_conv<3,64,32,32,48,64><<<(tiles+7)/8, 256, 0, stream>>>(x016, encw16[0], enc_b[0], a1); }
  { int tiles = (320*16*16/16)*(64/16);
    k_conv<32,32,64,16,512,512><<<(tiles+7)/8, 256, 0, stream>>>(a1, encw16[1], enc_b[1], a2); }
  { int tiles = (320*8*8/16)*(128/16);
    k_conv<64,16,128,8,1024,1024><<<(tiles+7)/8, 256, 0, stream>>>(a2, encw16[2], enc_b[2], a3); }
  { int tiles = (320*4*4/16)*(256/16);
    k_conv<128,8,256,4,2048,2048><<<(tiles+7)/8, 256, 0, stream>>>(a3, encw16[3], enc_b[3], feats); }

  // ig = feats @ Wih^T + gru_b   (320 x 4096 x 768)
  { int tiles = (320/16)*(768/16);
    k_linear<<<(tiles + 7) / 8, 256, 0, stream>>>(feats, wih16, gru_b, ig, 320, 4096, 768, 0); }

  k_gru<<<1, 256, 0, stream>>>(ig, whh16, gru_bn, muw16, mu_b, lvw16, lv_b, eps,
                               hg, muo, lvo, z0, ztraj);

  k_ode<<<1, 256, 0, stream>>>(z0, vfw16[0], vf_b[0], vfw16[1], vf_b[1],
                               vfw16[2], vf_b[2], vfw16[3], vf_b[3], ztraj);

  // decoder linear: 640 x 64 x 2048 -> d0 (f16)
  { int tiles = (640/16)*(2048/16);
    k_linear<<<(tiles + 7) / 8, 256, 0, stream>>>(ztraj, declw16, decl_b, d0, 640, 64, 2048, 1); }

  // transposed convs
  { int tiles = (640*4*4/16)*(64/16);   dim3 g((tiles+7)/8, 4);
    k_convT<128,4,64,64,8,0><<<g, 256, 0, stream>>>(d0, decw16[0], dec_b[0], d1); }
  { int tiles = (640*8*8/16)*(32/16);   dim3 g((tiles+7)/8, 4);
    k_convT<64,8,32,32,16,0><<<g, 256, 0, stream>>>(d1, decw16[1], dec_b[1], d2); }
  { int tiles = (640*16*16/16)*(16/16); dim3 g((tiles+7)/8, 4);
    k_convT<32,16,16,16,32,0><<<g, 256, 0, stream>>>(d2, decw16[2], dec_b[2], d3); }
  { int tiles = (640*32*32/16)*(16/16); dim3 g((tiles+7)/8, 4);
    k_convT<16,32,3,16,64,1><<<g, 256, 0, stream>>>(d3, decw16[3], dec_b[3], d_out); }

  k_kl<<<1, 64, 0, stream>>>(muo, lvo, (float*)d_out + (size_t)32*20*64*64*3);

  (void)in_sizes; (void)n_in; (void)out_size; (void)ws_size;
}
